// SteeringAttention_60790967108195
// MI455X (gfx1250) — matrix operand order repaired and checked by reading, not compiled, not run on hardware
//
#include <hip/hip_runtime.h>
#include <hip/hip_bf16.h>
#include <math.h>

// ---------------------------------------------------------------------------
// SteeringAttention on MI455X (gfx1250): bf16 WMMA pipeline.
//   B=1, S=2048, NH=32, NKV=8, HD=128, HIDDEN=4096, GROUPS=4
// ---------------------------------------------------------------------------

typedef __attribute__((ext_vector_type(16))) __bf16 v16bf;
typedef __attribute__((ext_vector_type(8)))  float  v8f;
typedef int v4i_vs __attribute__((vector_size(16)));   // matches async builtin proto

#define WMMA_BF16(a, b, c) \
    __builtin_amdgcn_wmma_f32_16x16x32_bf16(false, (a), false, (b), (short)0, (c), false, false)

#if __has_builtin(__builtin_amdgcn_global_load_async_to_lds_b128) && \
    __has_builtin(__builtin_amdgcn_s_wait_asynccnt)
#define USE_ASYNC_LDS 1
#endif

#define AS1 __attribute__((address_space(1)))
#define AS3 __attribute__((address_space(3)))

__device__ __forceinline__ v8f v8f_zero() {
    v8f z;
#pragma unroll
    for (int e = 0; e < 8; ++e) z[e] = 0.0f;
    return z;
}

#ifdef USE_ASYNC_LDS
__device__ __forceinline__ void cp_async_b128(void* lds, const void* gaddr) {
    void* g = const_cast<void*>(gaddr);
    __builtin_amdgcn_global_load_async_to_lds_b128(
        (AS1 v4i_vs*)g, (AS3 v4i_vs*)lds, 0, 0);
}
#endif

// ---------------------------------------------------------------------------
// GEMM: C[M x N] = A[M x K] * W[N x K]^T  (fp32 in memory, bf16 WMMA, fp32 acc)
// Block tile 128x128, K-step 32, 256 threads = 8 waves (2 x 4 wave grid),
// each wave owns 64x32 = 4x2 subtiles of 16x16.
// Software pipeline: global->VGPR prefetch of tile k+1 overlaps WMMA on tile k;
// ping-pong LDS buffers give one barrier per K-step.
// ---------------------------------------------------------------------------
__global__ __launch_bounds__(256) void gemm_bf16_wmma(
    const float* __restrict__ A, const float* __restrict__ W,
    float* __restrict__ C, int M, int N, int K)
{
    __shared__ __bf16 As[2][128][32];
    __shared__ __bf16 Bs[2][128][32];

    const int tid  = threadIdx.x;
    const int lane = tid & 31;
    const int wave = tid >> 5;
    const int wm   = wave >> 2;      // 0..1  (64-row band)
    const int wn   = wave & 3;       // 0..3  (32-col band)
    const int half = lane >> 4;      // 0/1
    const int ln   = lane & 15;
    const int m0   = blockIdx.y * 128;
    const int n0   = blockIdx.x * 128;

    v8f acc[4][2];
#pragma unroll
    for (int i = 0; i < 4; ++i)
#pragma unroll
        for (int j = 0; j < 2; ++j) acc[i][j] = v8f_zero();

    // Per-thread staging coordinates: 128x32 tile as 1024 float4 chunks.
    int srow[4], scol[4];
    float4 pa[4], pb[4];
#pragma unroll
    for (int t = 0; t < 4; ++t) {
        int idx = tid + t * 256;
        srow[t] = idx >> 3;
        scol[t] = (idx & 7) << 2;
        pa[t] = *(const float4*)(A + (size_t)(m0 + srow[t]) * K + scol[t]);
        pb[t] = *(const float4*)(W + (size_t)(n0 + srow[t]) * K + scol[t]);
    }

    int p = 0;
    for (int k0 = 0; k0 < K; k0 += 32) {
        // Commit staged registers into LDS buffer p (fp32 -> bf16).
#pragma unroll
        for (int t = 0; t < 4; ++t) {
            As[p][srow[t]][scol[t] + 0] = (__bf16)pa[t].x;
            As[p][srow[t]][scol[t] + 1] = (__bf16)pa[t].y;
            As[p][srow[t]][scol[t] + 2] = (__bf16)pa[t].z;
            As[p][srow[t]][scol[t] + 3] = (__bf16)pa[t].w;
            Bs[p][srow[t]][scol[t] + 0] = (__bf16)pb[t].x;
            Bs[p][srow[t]][scol[t] + 1] = (__bf16)pb[t].y;
            Bs[p][srow[t]][scol[t] + 2] = (__bf16)pb[t].z;
            Bs[p][srow[t]][scol[t] + 3] = (__bf16)pb[t].w;
        }
        __syncthreads();

        // Prefetch next K-tile while this tile's WMMAs run.
        if (k0 + 32 < K) {
#pragma unroll
            for (int t = 0; t < 4; ++t) {
                pa[t] = *(const float4*)(A + (size_t)(m0 + srow[t]) * K + (k0 + 32) + scol[t]);
                pb[t] = *(const float4*)(W + (size_t)(n0 + srow[t]) * K + (k0 + 32) + scol[t]);
            }
        }

        const __bf16 (*Asp)[32] = As[p];
        const __bf16 (*Bsp)[32] = Bs[p];

        // B fragments (ISA B layout: N = lane%16, K = (e/8)*16 + (lane/16)*8 + e%8)
        v16bf bfr[2];
#pragma unroll
        for (int j = 0; j < 2; ++j) {
            int nB = wn * 32 + j * 16;
#pragma unroll
            for (int e = 0; e < 16; ++e)
                bfr[j][e] = Bsp[nB + ln][((e >> 3) << 4) + (half << 3) + (e & 7)];
        }
        // A fragments (ISA A layout) + WMMA
#pragma unroll
        for (int i = 0; i < 4; ++i) {
            int mB = wm * 64 + i * 16;
            v16bf a;
#pragma unroll
            for (int e = 0; e < 16; ++e) {
                int r = e >> 1, sub = e & 1;
                int kk = ((r >> 2) << 4) + (half << 3) + ((r & 3) << 1) + sub;
                a[e] = Asp[mB + ln][kk];
            }
            acc[i][0] = WMMA_BF16(a, bfr[0], acc[i][0]);
            acc[i][1] = WMMA_BF16(a, bfr[1], acc[i][1]);
        }
        p ^= 1;
    }

    // C layout: VGPR r -> row r + 8*half, col = lane%16
#pragma unroll
    for (int i = 0; i < 4; ++i)
#pragma unroll
        for (int j = 0; j < 2; ++j)
#pragma unroll
            for (int r = 0; r < 8; ++r) {
                int row = m0 + wm * 64 + i * 16 + r + (half << 3);
                int col = n0 + wn * 32 + j * 16 + ln;
                C[(size_t)row * N + col] = acc[i][j][r];
            }
}

// ---------------------------------------------------------------------------
// RoPE + steering scales + bf16 head-major repack + per-key logit bonus.
//   bonus[s] = clip(0.5*prior, -3, 3) + log(1 + 0.3*prior)   (softmax folding)
// ---------------------------------------------------------------------------
__global__ __launch_bounds__(128) void rope_scale_kernel(
    const float* __restrict__ qkv32, const float* __restrict__ prior,
    __bf16* __restrict__ qb, __bf16* __restrict__ kb, __bf16* __restrict__ vb,
    float* __restrict__ bonus)
{
    const int s = blockIdx.x;      // position 0..2047
    const int d = threadIdx.x;     // 0..127
    const int i = d & 63;
    const float inv_freq = __expf(-((float)(2 * i) / 128.0f) * __logf(10000.0f));
    const float ang = (float)s * inv_freq;
    const float c  = __cosf(ang);
    const float sn = __sinf(ang);

    const float* q32 = qkv32;
    const float* k32 = qkv32 + (size_t)2048 * 4096;
    const float* v32 = k32   + (size_t)2048 * 1024;

    const float p     = prior[s];
    const float gamma = fminf(fmaxf(1.0f + 0.1f * p, 0.9f), 1.1f);
    const float eta   = fminf(fmaxf(1.0f + 0.1f * p, 0.9f), 1.1f);
    const int dro = (d < 64) ? d + 64 : d - 64;

#pragma unroll 4
    for (int h = 0; h < 32; ++h) {
        float x  = q32[(size_t)s * 4096 + h * 128 + d];
        float xo = q32[(size_t)s * 4096 + h * 128 + dro];
        float rot = (d < 64) ? -xo : xo;
        qb[((size_t)h * 2048 + s) * 128 + d] = (__bf16)(x * c + rot * sn);
    }
#pragma unroll 2
    for (int h = 0; h < 8; ++h) {
        float x  = k32[(size_t)s * 1024 + h * 128 + d];
        float xo = k32[(size_t)s * 1024 + h * 128 + dro];
        float rot = (d < 64) ? -xo : xo;
        kb[((size_t)h * 2048 + s) * 128 + d] = (__bf16)((x * c + rot * sn) * gamma);
        vb[((size_t)h * 2048 + s) * 128 + d] =
            (__bf16)(v32[(size_t)s * 1024 + h * 128 + d] * eta);
    }
    if (d == 0) {
        float bias = fminf(fmaxf(0.5f * p, -3.0f), 3.0f);
        bonus[s] = bias + __logf(1.0f + 0.3f * p);
    }
}

// ---------------------------------------------------------------------------
// Flash attention, causal, GQA(4). Block = 128 thr (4 waves) on one head,
// 64 query rows (16 per wave). Key blocks of 32, double-buffered in LDS via
// CDNA5 async global->LDS copies (ASYNCcnt) when available.
// ---------------------------------------------------------------------------
__global__ __launch_bounds__(128) void attn_kernel(
    const __bf16* __restrict__ qb, const __bf16* __restrict__ kb,
    const __bf16* __restrict__ vb, const float* __restrict__ bonus,
    float* __restrict__ O)
{
    __shared__ __bf16 Ks[2][32][128];
    __shared__ __bf16 Vs[2][32][128];
    __shared__ __bf16 Pst[4][16][32];

    const int tid  = threadIdx.x;
    const int lane = tid & 31;
    const int wave = tid >> 5;
    const int half = lane >> 4;
    const int ln   = lane & 15;
    const int h    = blockIdx.y;           // head
    const int kh   = h >> 2;               // kv head (GQA)
    const int qr0  = blockIdx.x * 64 + wave * 16;

    // Q fragments: 16 rows x 128 d as 4 K=32 A-fragments (loaded once).
    v16bf qf[4];
#pragma unroll
    for (int kk = 0; kk < 4; ++kk)
#pragma unroll
        for (int e = 0; e < 16; ++e) {
            int r = e >> 1, sub = e & 1;
            int dd = kk * 32 + ((r >> 2) << 4) + (half << 3) + ((r & 3) << 1) + sub;
            qf[kk][e] = qb[((size_t)h * 2048 + qr0 + ln) * 128 + dd];
        }

    float rm[8], rl[8];
    v8f o[8];
#pragma unroll
    for (int r = 0; r < 8; ++r) { rm[r] = -1e30f; rl[r] = 0.0f; }
#pragma unroll
    for (int n = 0; n < 8; ++n) o[n] = v8f_zero();

    const float scl = 0.08838834764831845f;   // 1/sqrt(128)
    const int jmax = (blockIdx.x * 64 + 63) >> 5;

    // Per-thread staging chunks: 32x128 bf16 tile = 512 x (8 bf16), 4/thread.
    int crow[4], cch[4];
#pragma unroll
    for (int t = 0; t < 4; ++t) {
        int c = tid + t * 128;
        crow[t] = c >> 4;
        cch[t]  = (c & 15) << 3;
    }

#ifdef USE_ASYNC_LDS
    // Prologue: issue async copies for key block 0 into buffer 0.
#pragma unroll
    for (int t = 0; t < 4; ++t) {
        cp_async_b128(&Ks[0][crow[t]][cch[t]],
                      kb + ((size_t)kh * 2048 + crow[t]) * 128 + cch[t]);
        cp_async_b128(&Vs[0][crow[t]][cch[t]],
                      vb + ((size_t)kh * 2048 + crow[t]) * 128 + cch[t]);
    }
#endif

    for (int j = 0; j <= jmax; ++j) {
        const int p   = j & 1;
        const int kb0 = j << 5;
#ifdef USE_ASYNC_LDS
        if (j < jmax) {
            // Issue async copies for the next key block, then wait until only
            // those 8 remain outstanding => block j's data has landed.
            const int nb0 = (j + 1) << 5;
#pragma unroll
            for (int t = 0; t < 4; ++t) {
                cp_async_b128(&Ks[p ^ 1][crow[t]][cch[t]],
                              kb + ((size_t)kh * 2048 + nb0 + crow[t]) * 128 + cch[t]);
                cp_async_b128(&Vs[p ^ 1][crow[t]][cch[t]],
                              vb + ((size_t)kh * 2048 + nb0 + crow[t]) * 128 + cch[t]);
            }
            __builtin_amdgcn_s_wait_asynccnt(8);
        } else {
            __builtin_amdgcn_s_wait_asynccnt(0);
        }
#else
        // Synchronous fallback staging.
#pragma unroll
        for (int t = 0; t < 4; ++t) {
            *(uint4*)&Ks[p][crow[t]][cch[t]] =
                *(const uint4*)&kb[((size_t)kh * 2048 + kb0 + crow[t]) * 128 + cch[t]];
            *(uint4*)&Vs[p][crow[t]][cch[t]] =
                *(const uint4*)&vb[((size_t)kh * 2048 + kb0 + crow[t]) * 128 + cch[t]];
        }
#endif
        __syncthreads();

        if (kb0 <= qr0 + 15) {
            const __bf16 (*Kp)[128] = Ks[p];
            const __bf16 (*Vp)[128] = Vs[p];

            // S tile 16x32 = two 16x16 WMMA outputs over HD=128.
            v8f s0 = v8f_zero(), s1 = v8f_zero();
#pragma unroll
            for (int kk = 0; kk < 4; ++kk) {
                v16bf b0, b1;
#pragma unroll
                for (int e = 0; e < 16; ++e) {
                    int dd = kk * 32 + ((e >> 3) << 4) + (half << 3) + (e & 7);
                    b0[e] = Kp[ln][dd];
                    b1[e] = Kp[16 + ln][dd];
                }
                s0 = WMMA_BF16(qf[kk], b0, s0);
                s1 = WMMA_BF16(qf[kk], b1, s1);
            }

            const float bon0 = bonus[kb0 + ln];
            const float bon1 = bonus[kb0 + 16 + ln];
            const int   c0   = kb0 + ln;
            const int   c1   = kb0 + 16 + ln;

#pragma unroll
            for (int r = 0; r < 8; ++r) {
                int row = qr0 + r + (half << 3);
                float e0 = s0[r] * scl + bon0; if (c0 > row) e0 = -1e9f;
                float e1 = s1[r] * scl + bon1; if (c1 > row) e1 = -1e9f;

                float mcur = fmaxf(e0, e1);
#pragma unroll
                for (int off = 1; off < 16; off <<= 1)
                    mcur = fmaxf(mcur, __shfl_xor(mcur, off, 32));
                float mnew = fmaxf(rm[r], mcur);
                float corr = __expf(rm[r] - mnew);
                rm[r] = mnew;

                float p0 = __expf(e0 - mnew);
                float p1 = __expf(e1 - mnew);
                float ps = p0 + p1;
#pragma unroll
                for (int off = 1; off < 16; off <<= 1)
                    ps += __shfl_xor(ps, off, 32);
                rl[r] = rl[r] * corr + ps;
#pragma unroll
                for (int n = 0; n < 8; ++n) o[n][r] *= corr;

                Pst[wave][r + (half << 3)][ln]      = (__bf16)p0;
                Pst[wave][r + (half << 3)][16 + ln] = (__bf16)p1;
            }

            // P as A-fragment (transpose via per-wave LDS slice).
            v16bf pa;
#pragma unroll
            for (int e = 0; e < 16; ++e) {
                int r = e >> 1, sub = e & 1;
                int kk = ((r >> 2) << 4) + (half << 3) + ((r & 3) << 1) + sub;
                pa[e] = Pst[wave][ln][kk];
            }
            // O += P @ V : 8 d-subtiles of 16.
#pragma unroll
            for (int n = 0; n < 8; ++n) {
                v16bf bv;
#pragma unroll
                for (int e = 0; e < 16; ++e)
                    bv[e] = Vp[((e >> 3) << 4) + (half << 3) + (e & 7)][n * 16 + ln];
                o[n] = WMMA_BF16(pa, bv, o[n]);
            }
        }
        __syncthreads();   // protect buffer p^1 before it is overwritten
    }

    // Normalize and write O row-major [S][4096] fp32.
#pragma unroll
    for (int n = 0; n < 8; ++n)
#pragma unroll
        for (int r = 0; r < 8; ++r) {
            int row = qr0 + r + (half << 3);
            int col = h * 128 + n * 16 + ln;
            O[(size_t)row * 4096 + col] = o[n][r] / rl[r];
        }
}

// ---------------------------------------------------------------------------
// Host-side orchestration.
// ---------------------------------------------------------------------------
extern "C" void kernel_launch(void* const* d_in, const int* in_sizes, int n_in,
                              void* d_out, int out_size, void* d_ws, size_t ws_size,
                              hipStream_t stream) {
    (void)in_sizes; (void)n_in; (void)out_size; (void)ws_size;
    const float* hidden = (const float*)d_in[0];
    // d_in[1] attention_mask: causal mask computed inline; d_in[2] position_ids: arange.
    const float* prior  = (const float*)d_in[3];
    const float* Wq     = (const float*)d_in[4];
    const float* Wk     = (const float*)d_in[5];
    const float* Wv     = (const float*)d_in[6];
    const float* Wo     = (const float*)d_in[7];
    float* out = (float*)d_out;

    // Workspace layout (~76 MB).
    float* q32 = (float*)d_ws;                          // 2048*4096 fp32
    float* k32 = q32 + (size_t)2048 * 4096;             // 2048*1024 fp32
    float* v32 = k32 + (size_t)2048 * 1024;             // 2048*1024 fp32
    __bf16* qb = (__bf16*)(v32 + (size_t)2048 * 1024);  // [32][2048][128] bf16
    __bf16* kb = qb + (size_t)32 * 2048 * 128;          // [8][2048][128] bf16
    __bf16* vb = kb + (size_t)8  * 2048 * 128;          // [8][2048][128] bf16
    float* bonus = (float*)(vb + (size_t)8 * 2048 * 128);
    float* Obuf  = q32;  // reuse q fp32 region for attention output

    // 1-3: Q/K/V projections (C = X @ W^T) with bf16 WMMA.
    gemm_bf16_wmma<<<dim3(4096 / 128, 2048 / 128), 256, 0, stream>>>(
        hidden, Wq, q32, 2048, 4096, 4096);
    gemm_bf16_wmma<<<dim3(1024 / 128, 2048 / 128), 256, 0, stream>>>(
        hidden, Wk, k32, 2048, 1024, 4096);
    gemm_bf16_wmma<<<dim3(1024 / 128, 2048 / 128), 256, 0, stream>>>(
        hidden, Wv, v32, 2048, 1024, 4096);

    // 4: RoPE + steering scales + bf16 repack + logit bonus table.
    rope_scale_kernel<<<2048, 128, 0, stream>>>(q32, prior, qb, kb, vb, bonus);

    // 5: causal flash attention (32 q-groups of 64 rows x 32 heads).
    attn_kernel<<<dim3(2048 / 64, 32), 128, 0, stream>>>(qb, kb, vb, bonus, Obuf);

    // 6: output projection.
    gemm_bf16_wmma<<<dim3(4096 / 128, 2048 / 128), 256, 0, stream>>>(
        Obuf, Wo, out, 2048, 4096, 4096);
}
